// MultiSelfAttention_50173807952733
// MI455X (gfx1250) — matrix operand order repaired and checked by reading, not compiled, not run on hardware
//
#include <hip/hip_runtime.h>

// ---------------------------------------------------------------------------
// Multi-head self-attention for MI455X (gfx1250), wave32 + WMMA bf16.
// B=8, P=1024, D=1024, H=16, N=64.
//   k1: QKV projection GEMM -> q,k bf16 [b][h][p][64], v bf16 TRANSPOSED [b][h][64][p]
//   k2: flash attention (barrier-free inner loop; K/V fragments straight from
//       global; wave-private P~ tile in LDS; exp2-domain online softmax)
//   k3: output projection GEMM -> f32 out
// All matmuls: v_wmma_f32_16x16x32_bf16 (f32 accumulate).
// Workspace: 3 * 16MB (q,k,vT bf16) + 32MB (concat f32) = 80MB.
// ---------------------------------------------------------------------------

typedef __attribute__((ext_vector_type(16))) __bf16 v16bf;
typedef __attribute__((ext_vector_type(8)))  __bf16 v8bf;
typedef __attribute__((ext_vector_type(8)))  float  v8f;
typedef __attribute__((ext_vector_type(4)))  float  v4f;

#define WMMA_BF16(a, b, c) \
  __builtin_amdgcn_wmma_f32_16x16x32_bf16(false, (a), false, (b), (short)0, (c), false, false)

union AFrag { v16bf v; v8bf h[2]; };

// ---------------------------------------------------------------------------
// Shared 64x64 (K-step 64) bf16 GEMM core with register-prefetch pipeline.
// A: f32 row-major, lda = 1024. B: f32, element Bp[k*ldb + n] for n in [0,64).
// 256 threads = 8 waves (waveM 0..3 x waveN 0..1); each wave owns 16x32 of C.
// ---------------------------------------------------------------------------
__device__ __forceinline__ void gemm_core_64x64(
    const float* __restrict__ A, const float* __restrict__ Bp, size_t ldb,
    int mBase, __bf16* As, __bf16* Bs, v8f& c0, v8f& c1)
{
  const int tid  = threadIdx.x;
  const int lane = tid & 31;
  const int wid  = tid >> 5;
  const int hl   = lane >> 4;      // wave half
  const int ln   = lane & 15;
  const int waveM = wid & 3;
  const int waveN = wid >> 2;

  const int arow  = tid >> 2;            // A stage: 4 threads/row, 16 k each
  const int akseg = (tid & 3) * 16;
  const int bn    = tid & 63;            // B stage: col n, 16 k each (transposed)
  const int bkseg = (tid >> 6) * 16;

  float ar[16], br[16];
  // Preload K-step 0 into registers.
  {
    const float* ap = A + (size_t)(mBase + arow) * 1024 + akseg;
    #pragma unroll
    for (int q = 0; q < 4; ++q) {
      v4f t = *(const v4f*)(ap + 4 * q);
      #pragma unroll
      for (int j = 0; j < 4; ++j) ar[4 * q + j] = t[j];
    }
    #pragma unroll
    for (int j = 0; j < 16; ++j) br[j] = Bp[(size_t)(bkseg + j) * ldb + bn];
  }

  for (int k0 = 0; k0 < 1024; k0 += 64) {
    __syncthreads();
    // Commit staged registers to LDS (hardware f32->bf16, 16B stores).
    {
      v8bf p0, p1;
      #pragma unroll
      for (int j = 0; j < 8; ++j) { p0[j] = (__bf16)ar[j]; p1[j] = (__bf16)ar[8 + j]; }
      *(v8bf*)&As[arow * 64 + akseg]     = p0;
      *(v8bf*)&As[arow * 64 + akseg + 8] = p1;
      #pragma unroll
      for (int j = 0; j < 8; ++j) { p0[j] = (__bf16)br[j]; p1[j] = (__bf16)br[8 + j]; }
      *(v8bf*)&Bs[bn * 64 + bkseg]     = p0;
      *(v8bf*)&Bs[bn * 64 + bkseg + 8] = p1;
    }
    __syncthreads();

    // Issue next K-step's global loads early (overlap with WMMA below).
    if (k0 + 64 < 1024) {
      const float* ap = A + (size_t)(mBase + arow) * 1024 + k0 + 64 + akseg;
      #pragma unroll
      for (int q = 0; q < 4; ++q) {
        v4f t = *(const v4f*)(ap + 4 * q);
        #pragma unroll
        for (int j = 0; j < 4; ++j) ar[4 * q + j] = t[j];
      }
      #pragma unroll
      for (int j = 0; j < 16; ++j)
        br[j] = Bp[(size_t)(k0 + 64 + bkseg + j) * ldb + bn];
    }

    // Fragments (contiguous k = 16*half + s in A and B) + 4 WMMA per K-step.
    AFrag a0, a1;
    const int r = waveM * 16 + ln;
    a0.h[0] = *(const v8bf*)&As[r * 64 + hl * 16];
    a0.h[1] = *(const v8bf*)&As[r * 64 + 8 + hl * 16];
    a1.h[0] = *(const v8bf*)&As[r * 64 + 32 + hl * 16];
    a1.h[1] = *(const v8bf*)&As[r * 64 + 40 + hl * 16];
    const int n0 = waveN * 32 + ln;
    v16bf b00 = *(const v16bf*)&Bs[n0 * 64 + hl * 16];
    v16bf b01 = *(const v16bf*)&Bs[n0 * 64 + 32 + hl * 16];
    v16bf b10 = *(const v16bf*)&Bs[(n0 + 16) * 64 + hl * 16];
    v16bf b11 = *(const v16bf*)&Bs[(n0 + 16) * 64 + 32 + hl * 16];
    c0 = WMMA_BF16(a0.v, b00, c0);
    c0 = WMMA_BF16(a1.v, b01, c0);
    c1 = WMMA_BF16(a0.v, b10, c1);
    c1 = WMMA_BF16(a1.v, b11, c1);
  }
}

// ---------------------------------------------------------------------------
// Kernel 1: QKV projection. One 64-col block = one (h, a) panel of w_qkv.
// q,k stored [b][h][p][64]; v stored transposed [b][h][64][p] so attention
// can load V B-fragments contiguously from global memory.
// ---------------------------------------------------------------------------
__global__ __launch_bounds__(256) void qkv_gemm_kernel(
    const float* __restrict__ X, const float* __restrict__ W,
    __bf16* __restrict__ qb, __bf16* __restrict__ kb, __bf16* __restrict__ vt)
{
  __shared__ __bf16 As[64 * 64];
  __shared__ __bf16 Bs[64 * 64];

  const int mBase = blockIdx.x * 64;
  const int ha    = blockIdx.y;                 // h*3 + a
  const float* Wp = W + (size_t)ha * 1024 * 64;

  v8f c0 = {}; v8f c1 = {};
  gemm_core_64x64(X, Wp, 64, mBase, As, Bs, c0, c1);

  const int lane = threadIdx.x & 31;
  const int wid  = threadIdx.x >> 5;
  const int hl   = lane >> 4, ln = lane & 15;
  const int waveM = wid & 3, waveN = wid >> 2;
  const int hh = ha / 3, aa = ha % 3;
  const int nc = waveN * 32 + ln;

  #pragma unroll
  for (int i = 0; i < 8; ++i) {
    int row = mBase + waveM * 16 + i + hl * 8;    // C layout: row = i + half*8
    int b = row >> 10, p = row & 1023;
    if (aa == 2) {  // V transposed: [b][h][n][p]
      size_t base = ((size_t)b * 16 + hh) * 64;
      vt[(base + nc) * 1024 + p]      = (__bf16)c0[i];
      vt[(base + nc + 16) * 1024 + p] = (__bf16)c1[i];
    } else {        // Q / K: [b][h][p][n]
      __bf16* dst = (aa == 0) ? qb : kb;
      size_t base = (((size_t)b * 16 + hh) * 1024 + p) * 64;
      dst[base + nc]      = (__bf16)c0[i];
      dst[base + nc + 16] = (__bf16)c1[i];
    }
  }
}

// ---------------------------------------------------------------------------
// Kernel 2: flash attention per (b, h, 64-query tile). 4 waves x 16 queries.
// No block barriers in the key loop: K and V fragments come straight from
// global (both contiguous per lane), P~ staging is wave-private LDS.
// ---------------------------------------------------------------------------
__global__ __launch_bounds__(128) void attn_kernel(
    const __bf16* __restrict__ qb, const __bf16* __restrict__ kb,
    const __bf16* __restrict__ vt, float* __restrict__ oc)
{
  __shared__ __bf16 Ps[4][16 * 32];   // per-wave P~ tile [row][key_local]

  const int tid  = threadIdx.x;
  const int lane = tid & 31;
  const int w    = tid >> 5;
  const int hl   = lane >> 4;
  const int ln   = lane & 15;

  const int qt = blockIdx.x & 15;
  const int h  = (blockIdx.x >> 4) & 15;
  const int b  = blockIdx.x >> 8;
  const int bh = b * 16 + h;
  const __bf16* Q  = qb + (size_t)bh * 1024 * 64;
  const __bf16* K  = kb + (size_t)bh * 1024 * 64;
  const __bf16* VT = vt + (size_t)bh * 64 * 1024;
  const int qBase = qt * 64 + w * 16;

  // Q fragments (A 16x32 per K-step), held in registers the whole kernel.
  AFrag aq[2];
  #pragma unroll
  for (int kk = 0; kk < 2; ++kk) {
    const __bf16* qp = Q + (size_t)(qBase + ln) * 64 + kk * 32 + hl * 16;
    aq[kk].h[0] = *(const v8bf*)qp;
    aq[kk].h[1] = *(const v8bf*)(qp + 8);
  }

  float mrow[8], lrow[8];
  v8f o[4] = {{}, {}, {}, {}};
  #pragma unroll
  for (int i = 0; i < 8; ++i) { mrow[i] = -1e30f; lrow[i] = 0.f; }
  // exp2 domain: softmax_e(s/8) == softmax_2(s * 0.125 * log2(e))
  const float sl2 = 0.125f * 1.44269504f;

  for (int keyBase = 0; keyBase < 1024; keyBase += 32) {
    // Prefetch next key block of K and V (global_prefetch_b8, WGP scope).
    if (keyBase + 32 < 1024) {
      __builtin_prefetch(K + (size_t)(keyBase + 32 + lane) * 64, 0, 3);
      __builtin_prefetch(VT + (size_t)lane * 1024 + keyBase + 32, 0, 3);
      __builtin_prefetch(VT + (size_t)(lane + 32) * 1024 + keyBase + 32, 0, 3);
    }

    // S = Q * K^T for two 16-key sub-blocks (B-frag column = key).
    v8f s[2];
    #pragma unroll
    for (int t = 0; t < 2; ++t) {
      const int key = keyBase + t * 16 + ln;
      v8f acc = {};
      #pragma unroll
      for (int kk = 0; kk < 2; ++kk) {
        v16bf bk = *(const v16bf*)(K + (size_t)key * 64 + kk * 32 + hl * 16);
        acc = WMMA_BF16(aq[kk].v, bk, acc);
      }
      s[t] = acc;
    }

    // Online softmax; rows live in 16-lane halves -> butterfly over masks 1..8.
    float alpha[8];
    #pragma unroll
    for (int i = 0; i < 8; ++i) {
      float x0 = s[0][i] * sl2, x1 = s[1][i] * sl2;
      float rmax = fmaxf(x0, x1);
      #pragma unroll
      for (int msk = 1; msk < 16; msk <<= 1)
        rmax = fmaxf(rmax, __shfl_xor(rmax, msk, 32));
      float mnew = fmaxf(mrow[i], rmax);
      alpha[i] = exp2f(mrow[i] - mnew);
      float p0 = exp2f(x0 - mnew);
      float p1 = exp2f(x1 - mnew);
      float rs = p0 + p1;
      #pragma unroll
      for (int msk = 1; msk < 16; msk <<= 1)
        rs += __shfl_xor(rs, msk, 32);
      lrow[i] = lrow[i] * alpha[i] + rs;
      mrow[i] = mnew;
      const int row = i + hl * 8;
      Ps[w][row * 32 + ln]      = (__bf16)p0;
      Ps[w][row * 32 + 16 + ln] = (__bf16)p1;
      #pragma unroll
      for (int t = 0; t < 4; ++t) o[t][i] *= alpha[i];   // rescale O
    }
    __builtin_amdgcn_wave_barrier();   // intra-wave LDS ordering point

    // O += P~ (16x32) x V (32x64); V B-frags contiguous from global vT.
    AFrag pa;
    pa.h[0] = *(const v8bf*)&Ps[w][ln * 32 + hl * 16];
    pa.h[1] = *(const v8bf*)&Ps[w][ln * 32 + 8 + hl * 16];
    #pragma unroll
    for (int t = 0; t < 4; ++t) {
      v16bf bv = *(const v16bf*)(VT + (size_t)(t * 16 + ln) * 1024 + keyBase + hl * 16);
      o[t] = WMMA_BF16(pa.v, bv, o[t]);
    }
  }

  // Normalize and write f32 concat buffer [b][p][h*64+n].
  #pragma unroll
  for (int i = 0; i < 8; ++i) {
    float inv = 1.0f / lrow[i];
    int p = qBase + i + hl * 8;
    size_t base = ((size_t)b * 1024 + p) * 1024 + h * 64;
    #pragma unroll
    for (int t = 0; t < 4; ++t)
      oc[base + t * 16 + ln] = o[t][i] * inv;
  }
}

// ---------------------------------------------------------------------------
// Kernel 3: output projection (concat f32 8192x1024) x (w_o 1024x1024).
// ---------------------------------------------------------------------------
__global__ __launch_bounds__(256) void oproj_gemm_kernel(
    const float* __restrict__ Xc, const float* __restrict__ Wo,
    float* __restrict__ out)
{
  __shared__ __bf16 As[64 * 64];
  __shared__ __bf16 Bs[64 * 64];

  const int mBase = blockIdx.x * 64;
  const int cBase = blockIdx.y * 64;

  v8f c0 = {}; v8f c1 = {};
  gemm_core_64x64(Xc, Wo + cBase, 1024, mBase, As, Bs, c0, c1);

  const int lane = threadIdx.x & 31;
  const int wid  = threadIdx.x >> 5;
  const int hl   = lane >> 4, ln = lane & 15;
  const int waveM = wid & 3, waveN = wid >> 2;

  #pragma unroll
  for (int i = 0; i < 8; ++i) {
    int row = mBase + waveM * 16 + i + hl * 8;
    int col = cBase + waveN * 32 + ln;
    out[(size_t)row * 1024 + col]      = c0[i];
    out[(size_t)row * 1024 + col + 16] = c1[i];
  }
}

// ---------------------------------------------------------------------------
extern "C" void kernel_launch(void* const* d_in, const int* in_sizes, int n_in,
                              void* d_out, int out_size, void* d_ws, size_t ws_size,
                              hipStream_t stream)
{
  (void)in_sizes; (void)n_in; (void)out_size; (void)ws_size;
  const float* x    = (const float*)d_in[0];  // (8,1024,1024)
  const float* wqkv = (const float*)d_in[1];  // (16,3,1024,64)
  const float* wo   = (const float*)d_in[2];  // (1024,1024)
  float* out = (float*)d_out;                 // (8,1024,1024)

  const size_t QKV_ELEMS = (size_t)8 * 16 * 1024 * 64;   // 8M elements each
  char* ws = (char*)d_ws;
  __bf16* qb = (__bf16*)ws;                                  // 16 MB
  __bf16* kb = qb + QKV_ELEMS;                               // 16 MB
  __bf16* vt = kb + QKV_ELEMS;                               // 16 MB (transposed)
  float*  oc = (float*)(ws + 3 * QKV_ELEMS * sizeof(__bf16)); // 32 MB

  // 1) QKV projection: M=8192 rows, 48 panels of 64 cols.
  qkv_gemm_kernel<<<dim3(128, 48), 256, 0, stream>>>(x, wqkv, qb, kb, vt);
  // 2) Flash attention: 8*16 (b,h) x 16 query tiles = 2048 blocks.
  attn_kernel<<<2048, 128, 0, stream>>>(qb, kb, vt, oc);
  // 3) Output projection: M=8192, N=1024.
  oproj_gemm_kernel<<<dim3(128, 16), 256, 0, stream>>>(oc, wo, out);
}
